// M3GNetConv_120259084577
// MI455X (gfx1250) — matrix operand order repaired and checked by reading, not compiled, not run on hardware
//
#include <hip/hip_runtime.h>

// ---------------- problem constants ----------------
#define N_NODES 50000
#define N_EDGES 800000
#define NF 64
#define EF 64
#define DEG 9
#define CF 192

#define TILE_E 128     // edges per block
#define WAVES  8       // 8 waves * 16 edges = 128
#define CSTR   208     // concat row stride (bf16 elems), padded from 192
#define SSTR   72      // stage row stride (bf16 elems), padded from 64

// swizzled-weights layout in d_ws (bf16 element offsets)
#define OFF_EW1 0
#define OFF_EG1 12288
#define OFF_EW2 24576
#define OFF_EG2 28672
#define OFF_NW1 32768
#define OFF_NG1 45056
#define OFF_NW2 57344
#define OFF_NG2 61440
#define WS_ELEMS 65536

typedef __attribute__((ext_vector_type(16))) __bf16 v16bf;
typedef __attribute__((ext_vector_type(8)))  __bf16 v8bf;
typedef __attribute__((ext_vector_type(8)))  float  v8f;
typedef __attribute__((ext_vector_type(4)))  float  v4f;

__device__ __forceinline__ float siluf(float x) { return x / (1.f + __expf(-x)); }
__device__ __forceinline__ float sigmf(float x) { return 1.f / (1.f + __expf(-x)); }

__device__ __forceinline__ v16bf cat16(v8bf lo, v8bf hi) {
  v16bf r;
#pragma unroll
  for (int i = 0; i < 8; i++) { r[i] = lo[i]; r[i + 8] = hi[i]; }
  return r;
}

// A operand: row-major [16][K] in LDS; per-lane: m = lane&15, half = lane>>4
// elements 0..7  = K = kt*32 + half*8 + {0..7}
// elements 8..15 = K = kt*32 + 16 + half*8 + {0..7}
// (caller pre-adds half*8 into base pointer)
__device__ __forceinline__ v16bf ldsA(const __bf16* base, int kOff) {
  v8bf lo = *(const v8bf*)(base + kOff);
  v8bf hi = *(const v8bf*)(base + kOff + 16);
  return cat16(lo, hi);
}

// B operand: pre-swizzled fragment in global: lane holds 16 contiguous bf16
__device__ __forceinline__ v16bf ldgB(const __bf16* p) {
  v8bf lo = *(const v8bf*)p;
  v8bf hi = *(const v8bf*)(p + 8);
  return cat16(lo, hi);
}

#define WMMA_BF16(a, b, c) \
  __builtin_amdgcn_wmma_f32_16x16x32_bf16(false, (a), false, (b), (short)0, (c), false, false)

// One gated MLP (in=192, hid=64, out=64) on a 16-edge tile, one wave.
// out[nt][r] = silu(h2 + b2) * sigmoid(g2 + gb2), C-layout accumulators.
__device__ __forceinline__ void gated_mlp_tile(
    const __bf16* __restrict__ concatRow,  // concat + rowIdx*CSTR  (rowIdx = waveId*16 + (lane&15))
    const __bf16* __restrict__ ws, int oW1, int oG1, int oW2, int oG2,
    const float* __restrict__ b1, const float* __restrict__ gb1,
    const float* __restrict__ b2, const float* __restrict__ gb2,
    __bf16* __restrict__ hbuf, __bf16* __restrict__ gbuf,  // per-wave [16][SSTR]
    int lane, v8f out[4])
{
  const int half = lane >> 4;
  const int nlo = lane & 15;
  const int laneOff = lane * 16;
  const __bf16* aBase = concatRow + half * 8;

  v8f h[4] = {}; v8f g[4] = {};
#pragma unroll
  for (int kt = 0; kt < 6; kt++) {
    v16bf a = ldsA(aBase, kt * 32);
#pragma unroll
    for (int nt = 0; nt < 4; nt++) {
      v16bf bw = ldgB(ws + oW1 + (kt * 4 + nt) * 512 + laneOff);
      h[nt] = WMMA_BF16(a, bw, h[nt]);
      v16bf bg = ldgB(ws + oG1 + (kt * 4 + nt) * 512 + laneOff);
      g[nt] = WMMA_BF16(a, bg, g[nt]);
    }
  }
  // layer-1 epilogue: bias + silu, restage bf16 row-major for layer-2 A operand
#pragma unroll
  for (int nt = 0; nt < 4; nt++) {
    int n = nt * 16 + nlo;
    float bb = b1[n], gg = gb1[n];
#pragma unroll
    for (int r = 0; r < 8; r++) {
      int mrow = half * 8 + r;
      hbuf[mrow * SSTR + n] = (__bf16)siluf(h[nt][r] + bb);
      gbuf[mrow * SSTR + n] = (__bf16)siluf(g[nt][r] + gg);
    }
  }
  asm volatile("s_wait_dscnt 0" ::: "memory");

  v8f h2[4] = {}; v8f g2[4] = {};
  const __bf16* hA = hbuf + nlo * SSTR + half * 8;
  const __bf16* gA = gbuf + nlo * SSTR + half * 8;
#pragma unroll
  for (int kt = 0; kt < 2; kt++) {
    v16bf ah = ldsA(hA, kt * 32);
    v16bf ag = ldsA(gA, kt * 32);
#pragma unroll
    for (int nt = 0; nt < 4; nt++) {
      v16bf bw = ldgB(ws + oW2 + (kt * 4 + nt) * 512 + laneOff);
      h2[nt] = WMMA_BF16(ah, bw, h2[nt]);
      v16bf bg = ldgB(ws + oG2 + (kt * 4 + nt) * 512 + laneOff);
      g2[nt] = WMMA_BF16(ag, bg, g2[nt]);
    }
  }
#pragma unroll
  for (int nt = 0; nt < 4; nt++) {
    int n = nt * 16 + nlo;
    float bb = b2[n], gg = gb2[n];
#pragma unroll
    for (int r = 0; r < 8; r++)
      out[nt][r] = siluf(h2[nt][r] + bb) * sigmf(g2[nt][r] + gg);
  }
}

// ---------------- kernel 1: pre-swizzle weights to bf16 B-fragments ----------------
__global__ void swizzle_weights_kernel(
    const float* __restrict__ ew1, const float* __restrict__ eg1,
    const float* __restrict__ ew2, const float* __restrict__ eg2,
    const float* __restrict__ nw1, const float* __restrict__ ng1,
    const float* __restrict__ nw2, const float* __restrict__ ng2,
    __bf16* __restrict__ ws)
{
  int idx = blockIdx.x * 256 + threadIdx.x;  // 0..65535
  const float* srcm; int base;
  if      (idx < OFF_EG1) { srcm = ew1; base = OFF_EW1; }
  else if (idx < OFF_EW2) { srcm = eg1; base = OFF_EG1; }
  else if (idx < OFF_EG2) { srcm = ew2; base = OFF_EW2; }
  else if (idx < OFF_NW1) { srcm = eg2; base = OFF_EG2; }
  else if (idx < OFF_NG1) { srcm = nw1; base = OFF_NW1; }
  else if (idx < OFF_NW2) { srcm = ng1; base = OFF_NG1; }
  else if (idx < OFF_NG2) { srcm = nw2; base = OFF_NW2; }
  else                    { srcm = ng2; base = OFF_NG2; }
  int li   = idx - base;
  int frag = li >> 9;          // 512 elements per 32x16 fragment
  int lane = (li >> 4) & 31;
  int pos  = li & 15;
  int kt = frag >> 2;          // 4 n-tiles per k-tile
  int nt = frag & 3;
  int k = kt * 32 + (lane >> 4) * 8 + (pos & 7) + (pos >> 3) * 16;  // lanes 0-15: K 0..7,16..23, lanes 16-31: K 8..15,24..31
  int n = nt * 16 + (lane & 15);
  ws[idx] = (__bf16)srcm[k * 64 + n];
}

// ---------------- kernel 2: init node output with residual base ----------------
__global__ void init_nodes_kernel(const float* __restrict__ nf, float* __restrict__ out) {
  long i = (long)blockIdx.x * 256 + threadIdx.x;  // one float4 each
  if (i < (long)N_NODES * NF / 4) ((v4f*)out)[i] = ((const v4f*)nf)[i];
}

// ---------------- kernel 3: fused edge+node message passing ----------------
__global__ __launch_bounds__(256) void m3gnet_kernel(
    const float* __restrict__ nodef, const float* __restrict__ edgea,
    const float* __restrict__ edgew,
    const int* __restrict__ src, const int* __restrict__ dst,
    const float* __restrict__ eb1, const float* __restrict__ eb2,
    const float* __restrict__ egb1, const float* __restrict__ egb2,
    const float* __restrict__ nb1, const float* __restrict__ nb2,
    const float* __restrict__ ngb1, const float* __restrict__ ngb2,
    const float* __restrict__ elin, const float* __restrict__ nlin,
    const __bf16* __restrict__ ws,
    float* __restrict__ node_out, float* __restrict__ edge_out)
{
  extern __shared__ char smem[];
  __bf16* concatB = (__bf16*)smem;                         // [128][208] bf16
  __bf16* stage   = concatB + TILE_E * CSTR;               // 8 waves * 2 * [16][72] bf16
  float*  ewls    = (float*)(stage + WAVES * 2 * 16 * SSTR);  // [128][12] f32
  float*  linE    = ewls + TILE_E * 12;                    // [9][64]
  float*  linN    = linE + 576;                            // [9][64]
  float*  biasLds = linN + 576;                            // 8 * 64
  int*    sidx    = (int*)(biasLds + 512);                 // [128]
  int*    didx    = sidx + TILE_E;                         // [128]

  const int tid = threadIdx.x;
  const long eBase = (long)blockIdx.x * TILE_E;

  // -------- phase A0: edge endpoints --------
  if (tid < TILE_E) {
    long e = eBase + tid; if (e >= N_EDGES) e = N_EDGES - 1;
    sidx[tid] = src[e];
    didx[tid] = dst[e];
  }
  __syncthreads();

  // -------- phase A1: stage edge_weights rows, lin mats, biases --------
  for (int i = tid; i < TILE_E * DEG; i += 256) {
    int e = i / DEG, d = i - e * DEG;
    long ge = eBase + e; if (ge >= N_EDGES) ge = N_EDGES - 1;
    ewls[e * 12 + d] = edgew[ge * DEG + d];
  }
  for (int i = tid; i < DEG * 64; i += 256) { linE[i] = elin[i]; linN[i] = nlin[i]; }
  if (tid < 64) {
    biasLds[tid]       = eb1[tid];  biasLds[64 + tid]  = eb2[tid];
    biasLds[128 + tid] = egb1[tid]; biasLds[192 + tid] = egb2[tid];
    biasLds[256 + tid] = nb1[tid];  biasLds[320 + tid] = nb2[tid];
    biasLds[384 + tid] = ngb1[tid]; biasLds[448 + tid] = ngb2[tid];
  }

  // -------- phase A2: gather [vi | vj | edge_attr] rows -> bf16 concat --------
  for (int c = tid; c < TILE_E * 24; c += 256) {   // 24 chunks of 8 floats per edge
    int e = c / 24, part = c - e * 24;
    long ge = eBase + e; if (ge >= N_EDGES) ge = N_EDGES - 1;
    const float* sp;
    if (part < 8)       sp = nodef + (long)sidx[e] * NF + part * 8;
    else if (part < 16) sp = nodef + (long)didx[e] * NF + (part - 8) * 8;
    else                sp = edgea + ge * EF + (part - 16) * 8;
    v4f x0 = *(const v4f*)sp;
    v4f x1 = *(const v4f*)(sp + 4);
    v8bf o;
    o[0] = (__bf16)x0[0]; o[1] = (__bf16)x0[1]; o[2] = (__bf16)x0[2]; o[3] = (__bf16)x0[3];
    o[4] = (__bf16)x1[0]; o[5] = (__bf16)x1[1]; o[6] = (__bf16)x1[2]; o[7] = (__bf16)x1[3];
    *(v8bf*)(concatB + e * CSTR + part * 8) = o;
  }
  __syncthreads();

  // -------- phase B: per-wave 16-edge tiles --------
  const int waveId = tid >> 5, lane = tid & 31;
  const int half = lane >> 4, nlo = lane & 15;
  const int rowIdx = waveId * 16 + nlo;
  const __bf16* concatRow = concatB + rowIdx * CSTR;
  __bf16* hbuf = stage + (waveId * 2) * 16 * SSTR;
  __bf16* gbuf = hbuf + 16 * SSTR;

  v8f out[4];

  // ===== edge MLP + residual edge update =====
  gated_mlp_tile(concatRow, ws, OFF_EW1, OFF_EG1, OFF_EW2, OFF_EG2,
                 biasLds + 0, biasLds + 128, biasLds + 64, biasLds + 192,
                 hbuf, gbuf, lane, out);

  {
    float lv[4][9];
#pragma unroll
    for (int nt = 0; nt < 4; nt++) {
      int n = nt * 16 + nlo;
#pragma unroll
      for (int d = 0; d < DEG; d++) lv[nt][d] = linE[d * 64 + n];
    }
#pragma unroll
    for (int r = 0; r < 8; r++) {
      int mrow = half * 8 + r;
      int eL = waveId * 16 + mrow;
      long ge = eBase + eL;
      float w[9];
#pragma unroll
      for (int d = 0; d < DEG; d++) w[d] = ewls[eL * 12 + d];
#pragma unroll
      for (int nt = 0; nt < 4; nt++) {
        int n = nt * 16 + nlo;
        float ew = 0.f;
#pragma unroll
        for (int d = 0; d < DEG; d++) ew += w[d] * lv[nt][d];
        float old = (float)concatB[eL * CSTR + 128 + n];
        float nv = old + out[nt][r] * ew;
        if (ge < N_EDGES) edge_out[ge * 64 + n] = nv;
        concatB[eL * CSTR + 128 + n] = (__bf16)nv;   // node MLP sees updated edge attrs
      }
    }
  }
  asm volatile("s_wait_dscnt 0" ::: "memory");

  // ===== node MLP + scatter-add messages =====
  gated_mlp_tile(concatRow, ws, OFF_NW1, OFF_NG1, OFF_NW2, OFF_NG2,
                 biasLds + 256, biasLds + 384, biasLds + 320, biasLds + 448,
                 hbuf, gbuf, lane, out);

  {
    float lv[4][9];
#pragma unroll
    for (int nt = 0; nt < 4; nt++) {
      int n = nt * 16 + nlo;
#pragma unroll
      for (int d = 0; d < DEG; d++) lv[nt][d] = linN[d * 64 + n];
    }
#pragma unroll
    for (int r = 0; r < 8; r++) {
      int mrow = half * 8 + r;
      int eL = waveId * 16 + mrow;
      long ge = eBase + eL;
      if (ge >= N_EDGES) continue;
      float w[9];
#pragma unroll
      for (int d = 0; d < DEG; d++) w[d] = ewls[eL * 12 + d];
      long nrow = (long)sidx[eL] * NF;
#pragma unroll
      for (int nt = 0; nt < 4; nt++) {
        int n = nt * 16 + nlo;
        float ew = 0.f;
#pragma unroll
        for (int d = 0; d < DEG; d++) ew += w[d] * lv[nt][d];
        atomicAdd(node_out + nrow + n, out[nt][r] * ew);
      }
    }
  }
}

extern "C" void kernel_launch(void* const* d_in, const int* in_sizes, int n_in,
                              void* d_out, int out_size, void* d_ws, size_t ws_size,
                              hipStream_t stream) {
  const float* nodef = (const float*)d_in[0];
  const float* edgea = (const float*)d_in[1];
  const float* edgew = (const float*)d_in[2];
  const int*   eidx  = (const int*)d_in[3];
  const float* ew1  = (const float*)d_in[4];  const float* eb1  = (const float*)d_in[5];
  const float* ew2  = (const float*)d_in[6];  const float* eb2  = (const float*)d_in[7];
  const float* eg1  = (const float*)d_in[8];  const float* egb1 = (const float*)d_in[9];
  const float* eg2  = (const float*)d_in[10]; const float* egb2 = (const float*)d_in[11];
  const float* elin = (const float*)d_in[12];
  const float* nw1  = (const float*)d_in[13]; const float* nb1  = (const float*)d_in[14];
  const float* nw2  = (const float*)d_in[15]; const float* nb2  = (const float*)d_in[16];
  const float* ng1  = (const float*)d_in[17]; const float* ngb1 = (const float*)d_in[18];
  const float* ng2  = (const float*)d_in[19]; const float* ngb2 = (const float*)d_in[20];
  const float* nlin = (const float*)d_in[21];

  float* node_out = (float*)d_out;
  float* edge_out = node_out + (long)N_NODES * NF;
  __bf16* ws = (__bf16*)d_ws;

  swizzle_weights_kernel<<<WS_ELEMS / 256, 256, 0, stream>>>(
      ew1, eg1, ew2, eg2, nw1, ng1, nw2, ng2, ws);

  init_nodes_kernel<<<(N_NODES * NF / 4 + 255) / 256, 256, 0, stream>>>(nodef, node_out);

  size_t smem = (size_t)(TILE_E * CSTR + WAVES * 2 * 16 * SSTR) * 2   // bf16 regions
              + (size_t)(TILE_E * 12 + 576 + 576 + 512) * 4           // f32 regions
              + (size_t)(TILE_E * 2) * 4;                             // index regions
  m3gnet_kernel<<<(N_EDGES + TILE_E - 1) / TILE_E, 256, smem, stream>>>(
      nodef, edgea, edgew, eidx, eidx + N_EDGES,
      eb1, eb2, egb1, egb2, nb1, nb2, ngb1, ngb2,
      elin, nlin, ws, node_out, edge_out);
}
